// DecoderRNN_9938554323033
// MI455X (gfx1250) — matrix operand order repaired and checked by reading, not compiled, not run on hardware
//
#include <hip/hip_runtime.h>
#include <stdint.h>

#define BATCH  64
#define SEQ    32
#define INPUT  2048
#define EMBED  512
#define HIDDEN 512
#define VOCAB  10000

typedef __bf16 bf16;
typedef __attribute__((ext_vector_type(16))) __bf16 v16bf;
typedef __attribute__((ext_vector_type(8)))  float  v8f;

// ---------- helpers ----------

__device__ __forceinline__ unsigned short f2bf_rne(float f) {
  union { float f; unsigned u; } v; v.f = f;
  unsigned u = v.u;
  unsigned r = u + 0x7FFFu + ((u >> 16) & 1u);   // round-to-nearest-even
  return (unsigned short)(r >> 16);
}
__device__ __forceinline__ void store_bf(bf16* p, float f) {
  unsigned short s = f2bf_rne(f);
  __builtin_memcpy((void*)p, &s, 2);
}

// Packed-A layout (16x32 bf16 tile, per CDNA5 ISA 7.12.2):
// lane = half*16 + (m&15), element e: k = (e>>3)*16 + half*8 + (e&7)
__device__ __forceinline__ int packA_off(int m, int k, int ktiles) {
  int mt = m >> 4, kt = k >> 5;
  int mr = m & 15, kr = k & 31;
  int half = (kr >> 3) & 1;
  int e    = ((kr >> 4) << 3) | (kr & 7);
  int lane = (half << 4) | mr;
  return ((mt * ktiles + kt) << 9) + (lane << 4) + e;
}

__device__ __forceinline__ v16bf load_frag(const bf16* base, int tile) {
  int lane = threadIdx.x & 31;
  return *reinterpret_cast<const v16bf*>(base + (((size_t)tile) << 9) + (lane << 4));
}

__device__ __forceinline__ v8f wmma_bf16(v16bf a, v16bf b, v8f c) {
  return __builtin_amdgcn_wmma_f32_16x16x32_bf16(false, a, false, b, (short)0, c,
                                                 false, false);
}

__device__ __forceinline__ float sigmoidf_(float x) {
  return 1.0f / (1.0f + __expf(-x));
}

// ---------- packing kernels ----------

// Pack a row-major KxN fp32 matrix into bf16 WMMA B-fragments.
// B tile (32x16): lane = half*16 + (n&15), element e -> k = (e>>3)*16 + half*8 + (e&7).
__global__ void pack_b_kernel(const float* __restrict__ src, bf16* __restrict__ dst,
                              int K, int N) {
  int p = blockIdx.x * blockDim.x + threadIdx.x;
  if (p >= K * N) return;
  int e    = p & 15;
  int lane = (p >> 4) & 31;
  int tile = p >> 9;
  int ktiles = K >> 5;
  int nt = tile / ktiles;
  int kt = tile - nt * ktiles;
  int k = (kt << 5) + ((e >> 3) << 4) + ((lane >> 4) << 3) + (e & 7);
  int n = (nt << 4) + (lane & 15);
  store_bf(dst + p, src[k * N + n]);
}

// Pack a row-major MxK fp32 matrix into bf16 WMMA A-fragments.
__global__ void pack_a_kernel(const float* __restrict__ src, bf16* __restrict__ dst,
                              int M, int K) {
  int p = blockIdx.x * blockDim.x + threadIdx.x;
  if (p >= M * K) return;
  int e    = p & 15;
  int lane = (p >> 4) & 31;
  int tile = p >> 9;
  int ktiles = K >> 5;
  int mt = tile / ktiles;
  int kt = tile - mt * ktiles;
  int m = (mt << 4) + (lane & 15);
  int k = (kt << 5) + ((e >> 3) << 4) + ((lane >> 4) << 3) + (e & 7);
  store_bf(dst + p, src[m * K + k]);
}

// x_bf[t] (t=1..31) = emb[captions[:, t-1]] in packed-A bf16 layout.
__global__ void embed_pack_kernel(const int* __restrict__ captions,
                                  const float* __restrict__ emb,
                                  bf16* __restrict__ x_bf) {
  int idx = blockIdx.x * blockDim.x + threadIdx.x;   // over 31*64*512
  if (idx >= (SEQ - 1) * BATCH * EMBED) return;
  int e = idx & (EMBED - 1);
  int b = (idx >> 9) & (BATCH - 1);
  int t = (idx >> 15) + 1;                           // 1..31
  int tok = captions[b * SEQ + (t - 1)];
  float v = emb[tok * EMBED + e];
  store_bf(x_bf + (size_t)t * (BATCH * EMBED) + packA_off(b, e, EMBED / 32), v);
}

__global__ void init_state_kernel(bf16* __restrict__ h_zero, float* __restrict__ c_buf) {
  int i = blockIdx.x * blockDim.x + threadIdx.x;
  if (i < BATCH * HIDDEN) {
    unsigned short z = 0;
    __builtin_memcpy((void*)(h_zero + i), &z, 2);
    c_buf[i] = 0.0f;
  }
}

// ---------- projection: feat = features @ W_proj + b_proj -> x_bf[0] ----------
// M=64, K=2048, N=512. Grid: 16 blocks x 256 threads; tile = mt*32 + nt.
__global__ __launch_bounds__(256) void proj_wmma_kernel(
    const bf16* __restrict__ featA,   // packed A 64x2048 (ktiles=64)
    const bf16* __restrict__ WprojB,  // packed B 2048x512 (ktiles=64, ntiles=32)
    const float* __restrict__ b_proj,
    bf16* __restrict__ x0)            // packed A 64x512 destination
{
  int wave = threadIdx.x >> 5;
  int lane = threadIdx.x & 31;
  int tile = blockIdx.x * 8 + wave;   // 0..127
  int mt = tile >> 5, nt = tile & 31;
  int n = (nt << 4) + (lane & 15);
  float bias = b_proj[n];
  v8f acc;
  for (int r = 0; r < 8; ++r) acc[r] = bias;
  for (int kt = 0; kt < 64; ++kt) {
    v16bf a = load_frag(featA, mt * 64 + kt);
    v16bf b = load_frag(WprojB, nt * 64 + kt);
    acc = wmma_bf16(a, b, acc);
  }
  int half = lane >> 4;
  for (int r = 0; r < 8; ++r) {
    int m = (mt << 4) + (half << 3) + r;
    store_bf(x0 + packA_off(m, n, HIDDEN / 32), acc[r]);
  }
}

// ---------- one LSTM step ----------
// Grid: 8 WGs (64 hidden cols each) x 256 threads (8 waves).
// Each wave owns 2 (mt,nt) output positions and accumulates ALL FOUR gates
// for each position: one A-fragment load feeds 4 WMMAs (4 independent
// accumulators also break the WMMA->WMMA D->A dependency chain).
__global__ __launch_bounds__(256) void lstm_step_kernel(
    const bf16* __restrict__ xt,     // packed A 64x512
    const bf16* __restrict__ hprev,  // packed A 64x512
    const bf16* __restrict__ Wg,     // 4x packed B 512x512 (i,f,c,o)
    const bf16* __restrict__ Ug,     // 4x packed B 512x512
    const float* __restrict__ b_i, const float* __restrict__ b_f,
    const float* __restrict__ b_c, const float* __restrict__ b_o,
    float* __restrict__ c_buf,       // 64x512 row-major f32
    bf16* __restrict__ hout)         // packed A 64x512 (== hs[t])
{
  __shared__ float gates[4][64][64];   // 64KB pre-activations
  int wave = threadIdx.x >> 5;
  int lane = threadIdx.x & 31;
  int hid0 = blockIdx.x << 6;          // 0,64,...,448
  const size_t GSZ = (size_t)HIDDEN * HIDDEN;

  for (int pp = 0; pp < 2; ++pp) {
    int pos = wave + (pp << 3);        // 0..15
    int mt = pos >> 2;                 // 0..3 (batch tile)
    int nt = pos & 3;                  // 0..3 (hidden tile within this WG)
    int ncol = hid0 + (nt << 4) + (lane & 15);
    int ntg = (hid0 >> 4) + nt;        // global n-tile 0..31

    v8f acc[4];
    {
      float bi = b_i[ncol], bf = b_f[ncol], bc = b_c[ncol], bo = b_o[ncol];
      for (int r = 0; r < 8; ++r) {
        acc[0][r] = bi; acc[1][r] = bf; acc[2][r] = bc; acc[3][r] = bo;
      }
    }

    for (int kt = 0; kt < 16; ++kt) {
      v16bf a = load_frag(xt, mt * 16 + kt);
#pragma unroll
      for (int g = 0; g < 4; ++g) {
        v16bf b = load_frag(Wg + g * GSZ, ntg * 16 + kt);
        acc[g] = wmma_bf16(a, b, acc[g]);
      }
    }
    for (int kt = 0; kt < 16; ++kt) {
      v16bf a = load_frag(hprev, mt * 16 + kt);
#pragma unroll
      for (int g = 0; g < 4; ++g) {
        v16bf b = load_frag(Ug + g * GSZ, ntg * 16 + kt);
        acc[g] = wmma_bf16(a, b, acc[g]);
      }
    }

    int half = lane >> 4;
    int jn = (nt << 4) + (lane & 15);
#pragma unroll
    for (int g = 0; g < 4; ++g)
      for (int r = 0; r < 8; ++r) {
        int m = (mt << 4) + (half << 3) + r;
        gates[g][m][jn] = acc[g][r];
      }
  }
  __syncthreads();

  for (int ii = 0; ii < 16; ++ii) {
    int cell = (ii << 8) + threadIdx.x;  // 0..4095
    int m = cell >> 6;                   // batch row
    int j = cell & 63;                   // local hidden col
    float i_t = sigmoidf_(gates[0][m][j]);
    float f_t = sigmoidf_(gates[1][m][j]);
    float g_t = tanhf(gates[2][m][j]);
    float o_t = sigmoidf_(gates[3][m][j]);
    int hc = hid0 + j;
    float c_old = c_buf[m * HIDDEN + hc];
    float c_new = f_t * c_old + i_t * g_t;
    float h_new = o_t * tanhf(c_new);
    c_buf[m * HIDDEN + hc] = c_new;
    store_bf(hout + packA_off(m, hc, HIDDEN / 32), h_new);
  }
}

// ---------- FC head: out[b][t][v] = hs[t][b] @ W_fc + b_fc ----------
// M=2048 (t,b), K=512, N=10000. Each wave computes 1 M-tile x 5 N-tiles
// (625 = 125*5): one A-fragment load feeds 5 WMMAs. 16000 groups, 8/block.
__global__ __launch_bounds__(256) void fc_wmma_kernel(
    const bf16* __restrict__ hsA,   // 32 consecutive packed 64x512 A blocks
    const bf16* __restrict__ WfcB,  // packed B 512x10000 (ktiles=16, ntiles=625)
    const float* __restrict__ b_fc,
    float* __restrict__ out)
{
  int wave = threadIdx.x >> 5;
  int lane = threadIdx.x & 31;
  int grp = blockIdx.x * 8 + wave;    // 0..15999
  int mt = grp & 127;                 // 0..127
  int ng = grp >> 7;                  // 0..124 -> N-tiles ng*5 .. ng*5+4

  v8f acc[5];
#pragma unroll
  for (int j = 0; j < 5; ++j) {
    int n = ((ng * 5 + j) << 4) + (lane & 15);
    float bias = b_fc[n];
    for (int r = 0; r < 8; ++r) acc[j][r] = bias;
  }

  for (int kt = 0; kt < 16; ++kt) {
    v16bf a = load_frag(hsA, mt * 16 + kt);
#pragma unroll
    for (int j = 0; j < 5; ++j) {
      v16bf b = load_frag(WfcB, (ng * 5 + j) * 16 + kt);
      acc[j] = wmma_bf16(a, b, acc[j]);
    }
  }

  int half = lane >> 4;
#pragma unroll
  for (int j = 0; j < 5; ++j) {
    int n = ((ng * 5 + j) << 4) + (lane & 15);
    for (int r = 0; r < 8; ++r) {
      int m = (mt << 4) + (half << 3) + r;  // global row in (t,b) order
      int t = m >> 6;
      int b = m & 63;
      out[(size_t)b * (SEQ * VOCAB) + (size_t)t * VOCAB + n] = acc[j][r];
    }
  }
}

// ---------- host side ----------

extern "C" void kernel_launch(void* const* d_in, const int* in_sizes, int n_in,
                              void* d_out, int out_size, void* d_ws, size_t ws_size,
                              hipStream_t stream) {
  (void)in_sizes; (void)n_in; (void)out_size; (void)ws_size;

  const float* features = (const float*)d_in[0];
  const int*   captions = (const int*)d_in[1];
  const float* W_proj   = (const float*)d_in[2];
  const float* b_proj   = (const float*)d_in[3];
  const float* emb      = (const float*)d_in[4];
  const float* W_i = (const float*)d_in[5];
  const float* U_i = (const float*)d_in[6];
  const float* b_i = (const float*)d_in[7];
  const float* W_f = (const float*)d_in[8];
  const float* U_f = (const float*)d_in[9];
  const float* b_f = (const float*)d_in[10];
  const float* W_c = (const float*)d_in[11];
  const float* U_c = (const float*)d_in[12];
  const float* b_c = (const float*)d_in[13];
  const float* W_o = (const float*)d_in[14];
  const float* U_o = (const float*)d_in[15];
  const float* b_o = (const float*)d_in[16];
  const float* W_fc = (const float*)d_in[17];
  const float* b_fc = (const float*)d_in[18];

  char* ws = (char*)d_ws;
  size_t off = 0;
  auto carve = [&](size_t bytes) -> char* {
    char* p = ws + off;
    off += (bytes + 255) & ~(size_t)255;
    return p;
  };

  const size_t XTB = (size_t)BATCH * EMBED;           // 32768 elems per timestep
  bf16* x_bf   = (bf16*)carve(SEQ * XTB * 2);         // packed A inputs, time-major
  bf16* hs_bf  = (bf16*)carve(SEQ * XTB * 2);         // packed A hidden states
  bf16* h_zero = (bf16*)carve(XTB * 2);
  float* c_buf = (float*)carve(XTB * 4);
  bf16* featA  = (bf16*)carve((size_t)BATCH * INPUT * 2);
  bf16* WprojB = (bf16*)carve((size_t)INPUT * EMBED * 2);
  bf16* WgB    = (bf16*)carve((size_t)4 * HIDDEN * HIDDEN * 2);
  bf16* UgB    = (bf16*)carve((size_t)4 * HIDDEN * HIDDEN * 2);
  bf16* WfcB   = (bf16*)carve((size_t)HIDDEN * VOCAB * 2);

  const int TPB = 256;
  auto blocks = [](size_t n) { return (unsigned)((n + 255) / 256); };

  init_state_kernel<<<blocks(BATCH * HIDDEN), TPB, 0, stream>>>(h_zero, c_buf);

  pack_a_kernel<<<blocks((size_t)BATCH * INPUT), TPB, 0, stream>>>(features, featA, BATCH, INPUT);
  pack_b_kernel<<<blocks((size_t)INPUT * EMBED), TPB, 0, stream>>>(W_proj, WprojB, INPUT, EMBED);

  const size_t GSZ = (size_t)HIDDEN * HIDDEN;
  pack_b_kernel<<<blocks(GSZ), TPB, 0, stream>>>(W_i, WgB + 0 * GSZ, EMBED, HIDDEN);
  pack_b_kernel<<<blocks(GSZ), TPB, 0, stream>>>(W_f, WgB + 1 * GSZ, EMBED, HIDDEN);
  pack_b_kernel<<<blocks(GSZ), TPB, 0, stream>>>(W_c, WgB + 2 * GSZ, EMBED, HIDDEN);
  pack_b_kernel<<<blocks(GSZ), TPB, 0, stream>>>(W_o, WgB + 3 * GSZ, EMBED, HIDDEN);
  pack_b_kernel<<<blocks(GSZ), TPB, 0, stream>>>(U_i, UgB + 0 * GSZ, HIDDEN, HIDDEN);
  pack_b_kernel<<<blocks(GSZ), TPB, 0, stream>>>(U_f, UgB + 1 * GSZ, HIDDEN, HIDDEN);
  pack_b_kernel<<<blocks(GSZ), TPB, 0, stream>>>(U_c, UgB + 2 * GSZ, HIDDEN, HIDDEN);
  pack_b_kernel<<<blocks(GSZ), TPB, 0, stream>>>(U_o, UgB + 3 * GSZ, HIDDEN, HIDDEN);
  pack_b_kernel<<<blocks((size_t)HIDDEN * VOCAB), TPB, 0, stream>>>(W_fc, WfcB, HIDDEN, VOCAB);

  embed_pack_kernel<<<blocks((size_t)(SEQ - 1) * BATCH * EMBED), TPB, 0, stream>>>(
      captions, emb, x_bf);

  proj_wmma_kernel<<<16, TPB, 0, stream>>>(featA, WprojB, b_proj, x_bf);

  for (int t = 0; t < SEQ; ++t) {
    const bf16* hprev = (t == 0) ? h_zero : (hs_bf + (size_t)(t - 1) * XTB);
    lstm_step_kernel<<<HIDDEN / 64, TPB, 0, stream>>>(
        x_bf + (size_t)t * XTB, hprev, WgB, UgB,
        b_i, b_f, b_c, b_o, c_buf, hs_bf + (size_t)t * XTB);
  }

  // 128 M-tiles * 125 N-groups = 16000 wave-groups, 8 per block.
  fc_wmma_kernel<<<2000, TPB, 0, stream>>>(hs_bf, WfcB, b_fc, (float*)d_out);
}
